// QLSTM_65481071404941
// MI455X (gfx1250) — matrix operand order repaired and checked by reading, not compiled, not run on hardware
//
#include <hip/hip_runtime.h>
#include <math.h>

// ---------------------------------------------------------------------------
// Problem constants (reference: T=512, B=64, D=768, H=768)
// ---------------------------------------------------------------------------
#define TT 512
#define BB 64
#define DD 768
#define HH 768
#define DH 1536          // D+H
#define G4 3072          // 4*H
#define ROWS (TT*BB)     // 32768
#define NWG 24           // persistent workgroups (768/32 hidden-unit slices)

typedef __attribute__((ext_vector_type(16))) __bf16  v16bf;
typedef __attribute__((ext_vector_type(8)))  float   v8f;
typedef __attribute__((ext_vector_type(4))) unsigned int u32x4;
typedef __attribute__((ext_vector_type(8)))  int          i32x8;
typedef __attribute__((ext_vector_type(4)))  int          i32x4;

struct Frag32B { u32x4 lo, hi; };

#if defined(__has_builtin)
#if __has_builtin(__builtin_amdgcn_tensor_load_to_lds)
#define HAVE_TDM 1
#endif
#endif

// fp32 -> bf16 round-to-nearest-even
__device__ __forceinline__ unsigned short f2bf(float f) {
    unsigned int u = __builtin_bit_cast(unsigned int, f);
    u += 0x7fffu + ((u >> 16) & 1u);
    return (unsigned short)(u >> 16);
}

// A-fragment (16x32 bf16, row-major source, ld elements) per ISA layout:
// lane 0-15 : M=lane,    K = {k0..k0+7, k0+16..k0+23}
// lane 16-31: M=lane-16, K = {k0+8..k0+15, k0+24..k0+31}
__device__ __forceinline__ v16bf load_a_frag(const unsigned short* base,
                                             int row0, int k0, int ld, int lane) {
    int m  = row0 + (lane & 15);
    int kh = lane >> 4;
    const unsigned short* p = base + (size_t)m * ld + k0 + kh * 8;
    Frag32B t;
    t.lo = *reinterpret_cast<const u32x4*>(p);
    t.hi = *reinterpret_cast<const u32x4*>(p + 16);
    return __builtin_bit_cast(v16bf, t);
}

// B-fragment (32x16 bf16); source is W row-major [n][k] (B = W^T):
// lane holds K = {k0+h*8..+7, k0+16+h*8..+7}, h = lane>>4.
__device__ __forceinline__ v16bf load_b_frag(const unsigned short* base,
                                             int col0, int k0, int ld, int lane) {
    int n  = col0 + (lane & 15);
    int kh = lane >> 4;
    const unsigned short* p = base + (size_t)n * ld + k0 + kh * 8;
    Frag32B t;
    t.lo = *reinterpret_cast<const u32x4*>(p);
    t.hi = *reinterpret_cast<const u32x4*>(p + 16);
    return __builtin_bit_cast(v16bf, t);
}

__device__ __forceinline__ v8f wmma_bf16(v16bf a, v16bf b, v8f c) {
    return __builtin_amdgcn_wmma_f32_16x16x32_bf16(false, a, false, b,
                                                   (short)0, c, false, false);
}

__device__ __forceinline__ float sigmoidf_(float x) {
    return 1.0f / (1.0f + expf(-x));
}

#ifdef HAVE_TDM
// Tensor DMA descriptor per CDNA5 ISA ch.8: 2D tile, data_size=2B,
// groups 2/3 unused (<=2D tensors).  stride0/dims in elements.
// 6-arg builtin form: (g0, g1, g2, g3, g4, cpol)
__device__ __forceinline__ void tdm_load_2d_bf16(unsigned lds_addr_bytes,
                                                 const void* gaddr,
                                                 unsigned tile_d0, unsigned tile_d1,
                                                 unsigned tensor_d0, unsigned tensor_d1,
                                                 unsigned long long stride0) {
    unsigned long long ga = (unsigned long long)(size_t)gaddr;
    u32x4 g0;
    g0.x = 1u;                                         // count=1, user descriptor
    g0.y = lds_addr_bytes;                             // lds_addr [63:32]
    g0.z = (unsigned)ga;                               // global_addr [95:64]
    g0.w = (unsigned)(ga >> 32) | 0x80000000u;         // addr[56:32] | type=2
    i32x8 g1;
    g1[0] = (int)(1u << 16);                           // data_size=1 (2 bytes)
    g1[1] = (int)((tensor_d0 & 0xffffu) << 16);        // tensor_dim0[15:0]
    g1[2] = (int)((tensor_d0 >> 16) | ((tensor_d1 & 0xffffu) << 16));
    g1[3] = (int)((tensor_d1 >> 16) | (tile_d0 << 16));// tile_dim0
    g1[4] = (int)(tile_d1 & 0xffffu);                  // tile_dim1 (tile_dim2=0)
    g1[5] = (int)(stride0 & 0xffffffffu);              // tensor_dim0_stride[31:0]
    g1[6] = (int)((stride0 >> 32) & 0xffffu);          // stride0[47:32], stride1=0
    g1[7] = 0;
    i32x4 z4 = {0, 0, 0, 0};
    i32x8 z8 = {0, 0, 0, 0, 0, 0, 0, 0};
    __builtin_amdgcn_tensor_load_to_lds(g0, g1, z4, z4, z8, 0);
}
#endif

// ---------------------------------------------------------------------------
// Phase A1: x fp32 -> bf16
// ---------------------------------------------------------------------------
__global__ void cvt_x_kernel(const float* __restrict__ x,
                             unsigned short* __restrict__ xbf, int n) {
    int i = blockIdx.x * blockDim.x + threadIdx.x;
    if (i < n) xbf[i] = f2bf(x[i]);
}

// ---------------------------------------------------------------------------
// Phase A2: repack gate weights [H, D+H] fp32 -> Wx bf16 [4H][D], Wh bf16 [4H][H]
// ---------------------------------------------------------------------------
__global__ void repack_w_kernel(const float* __restrict__ Wf,
                                const float* __restrict__ Wi,
                                const float* __restrict__ Wg,
                                const float* __restrict__ Wo,
                                unsigned short* __restrict__ Wx,
                                unsigned short* __restrict__ Wh) {
    int id = blockIdx.x * blockDim.x + threadIdx.x;
    const int total = 4 * HH * DH;
    if (id >= total) return;
    int g   = id / (HH * DH);
    int rem = id - g * (HH * DH);
    int r   = rem / DH;
    int c   = rem - r * DH;
    const float* W = (g == 0) ? Wf : (g == 1) ? Wi : (g == 2) ? Wg : Wo;
    unsigned short v = f2bf(W[(size_t)r * DH + c]);
    if (c < DD) Wx[((size_t)g * HH + r) * DD + c]        = v;
    else        Wh[((size_t)g * HH + r) * HH + (c - DD)] = v;
}

__global__ void zero_bar_kernel(unsigned* __restrict__ bar, int n) {
    int i = blockIdx.x * blockDim.x + threadIdx.x;
    if (i < n) bar[i] = 0u;
}

// ---------------------------------------------------------------------------
// Phase B: hoisted input projection Zx[ROWS][G4] = x_bf @ Wx^T (fp32 out)
// ---------------------------------------------------------------------------
__global__ void __launch_bounds__(256)
xproj_gemm_kernel(const unsigned short* __restrict__ xbf,
                  const unsigned short* __restrict__ Wx,
                  float* __restrict__ Zx) {
    const int lane = threadIdx.x & 31;
    const int w    = threadIdx.x >> 5;
    const int row0 = blockIdx.y * 64 + (w & 3) * 16;
    const int nb0  = blockIdx.x * 128 + (w >> 2) * 64;

    v8f acc[4] = {};
    for (int k0 = 0; k0 < DD; k0 += 32) {
        v16bf a = load_a_frag(xbf, row0, k0, DD, lane);
#pragma unroll
        for (int i = 0; i < 4; ++i) {
            v16bf b = load_b_frag(Wx, nb0 + i * 16, k0, DD, lane);
            acc[i] = wmma_bf16(a, b, acc[i]);
        }
    }
    const int nl = lane & 15;
    const int mh = lane >> 4;
#pragma unroll
    for (int i = 0; i < 4; ++i) {
        int n = nb0 + i * 16 + nl;
#pragma unroll
        for (int v = 0; v < 8; ++v) {
            int m = row0 + mh * 8 + v;
            Zx[(size_t)m * G4 + n] = acc[i][v];
        }
    }
}

// ---------------------------------------------------------------------------
// Phase C: persistent recurrent kernel. 24 WGs (one per WGP), 8 waves each.
// Wh slice (4 gates x 32 units x 768, bf16 = 192KB) loaded to LDS ONCE via
// TDM; h tile (64x768 bf16 = 96KB) refilled to LDS via TDM each step after a
// grid-wide atomic barrier. c-state lives in registers for all 512 steps.
// wave w: batch rows (w&3)*16..+16, hidden cols (w>>2)*16..+16 of the slice.
// ---------------------------------------------------------------------------
typedef __attribute__((address_space(3))) unsigned short* lds_us_ptr;

__global__ void __launch_bounds__(256)
lstm_persistent_kernel(const float* __restrict__ Zx,
                       const unsigned short* __restrict__ Wh,
                       const float* __restrict__ bfv,
                       const float* __restrict__ biv,
                       const float* __restrict__ bgv,
                       const float* __restrict__ bov,
                       unsigned short* __restrict__ hb0,
                       unsigned short* __restrict__ hb1,
                       unsigned* __restrict__ bar,
                       float* __restrict__ out) {
    extern __shared__ unsigned short smem[];
    unsigned short* sWh = smem;                 // 4*32*768 elems = 196608 B
    unsigned short* sH  = smem + 4 * 32 * HH;   // 64*768  elems =  98304 B

    const int tid  = threadIdx.x;
    const int lane = tid & 31;
    const int w    = tid >> 5;
    const int j0   = blockIdx.x * 32;           // this WG's hidden-unit slice
    const int row0 = (w & 3) * 16;              // batch rows for this wave
    const int colL = ((w >> 2) & 1) * 16;       // local cols within slice

    const unsigned ldsWh = (unsigned)(size_t)(lds_us_ptr)sWh;
    const unsigned ldsH  = (unsigned)(size_t)(lds_us_ptr)sH;
    (void)ldsWh; (void)ldsH;

    // h0 = 0 in LDS
    {
        u32x4 z4 = {0u, 0u, 0u, 0u};
        u32x4* p = reinterpret_cast<u32x4*>(sH);
        for (int i = tid; i < (BB * HH * 2) / 16; i += 256) p[i] = z4;
    }

    // Load this WG's Wh slice into LDS once (4 gate row-blocks of 32x768)
#ifdef HAVE_TDM
    if (w == 0) {
#pragma unroll
        for (int g = 0; g < 4; ++g) {
            tdm_load_2d_bf16(ldsWh + (unsigned)g * 32 * HH * 2,
                             Wh + ((size_t)g * HH + j0) * HH,
                             HH, 32, HH, 32, (unsigned long long)HH);
        }
        __builtin_amdgcn_s_wait_tensorcnt(0);
    }
#else
    {
        u32x4* dst = reinterpret_cast<u32x4*>(sWh);
        for (int g = 0; g < 4; ++g) {
            const u32x4* src = reinterpret_cast<const u32x4*>(
                Wh + ((size_t)g * HH + j0) * HH);
            for (int i = tid; i < (32 * HH * 2) / 16; i += 256)
                dst[g * (32 * HH / 8) + i] = src[i];
        }
    }
#endif
    __syncthreads();

    const int nl = lane & 15;
    const int mh = lane >> 4;
    const int j  = j0 + colL + nl;              // global hidden index
    const float bF = bfv[j], bI = biv[j], bG = bgv[j], bO = bov[j];

    float creg[8] = {0.f, 0.f, 0.f, 0.f, 0.f, 0.f, 0.f, 0.f};

    for (int t = 0; t < TT; ++t) {
        v8f acc[4] = {};
        for (int k0 = 0; k0 < HH; k0 += 32) {
            v16bf a = load_a_frag(sH, row0, k0, HH, lane);
#pragma unroll
            for (int g = 0; g < 4; ++g) {
                v16bf b = load_b_frag(sWh + g * 32 * HH, colL, k0, HH, lane);
                acc[g] = wmma_bf16(a, b, acc[g]);
            }
        }

        unsigned short* hout = (t & 1) ? hb0 : hb1;   // holds h_{t+1}
#pragma unroll
        for (int v = 0; v < 8; ++v) {
            int b = row0 + mh * 8 + v;
            const float* zrow = Zx + ((size_t)t * BB + b) * G4;
            float zf = acc[0][v] + zrow[j]          + bF;
            float zi = acc[1][v] + zrow[HH + j]     + bI;
            float zg = acc[2][v] + zrow[2 * HH + j] + bG;
            float zo = acc[3][v] + zrow[3 * HH + j] + bO;

            float fg = sigmoidf_(zf);
            float ig = sigmoidf_(zi);
            float gg = tanhf(zg);
            float og = sigmoidf_(zo);

            float cn = fg * creg[v] + ig * gg;
            float hn = og * tanhf(cn);
            creg[v] = cn;

            hout[(size_t)b * HH + j] = f2bf(hn);
            out[((size_t)t * BB + b) * HH + j] = hn;
            if (t == TT - 1) {
                out[(size_t)TT * BB * HH + (size_t)b * HH + j]                    = hn; // hT
                out[(size_t)TT * BB * HH + (size_t)BB * HH + (size_t)b * HH + j]  = cn; // cT
            }
        }

        if (t < TT - 1) {
            __threadfence();           // h_{t+1} stores visible at agent scope
            __syncthreads();           // all waves fenced + done reading sH
            if (tid == 0) {
                atomicAdd(&bar[t], 1u);
                while (__hip_atomic_load(&bar[t], __ATOMIC_RELAXED,
                                         __HIP_MEMORY_SCOPE_AGENT) < NWG) {
                    __builtin_amdgcn_s_sleep(1);
                }
            }
            __syncthreads();
            // refill full h_{t+1} into LDS (broadcast copy, 96KB)
#ifdef HAVE_TDM
            if (w == 0) {
                tdm_load_2d_bf16(ldsH, hout, BB * HH, 1, BB * HH, 1,
                                 (unsigned long long)(BB * HH));
                __builtin_amdgcn_s_wait_tensorcnt(0);
            }
#else
            {
                const u32x4* src = reinterpret_cast<const u32x4*>(hout);
                u32x4* dst = reinterpret_cast<u32x4*>(sH);
                for (int i = tid; i < (BB * HH * 2) / 16; i += 256) dst[i] = src[i];
            }
#endif
            __syncthreads();
        }
    }
}

// ---------------------------------------------------------------------------
// Host launcher
// ---------------------------------------------------------------------------
extern "C" void kernel_launch(void* const* d_in, const int* in_sizes, int n_in,
                              void* d_out, int out_size, void* d_ws, size_t ws_size,
                              hipStream_t stream) {
    const float* x  = (const float*)d_in[0];
    const float* Wf = (const float*)d_in[1];
    const float* bf = (const float*)d_in[2];
    const float* Wi = (const float*)d_in[3];
    const float* bi = (const float*)d_in[4];
    const float* Wg = (const float*)d_in[5];
    const float* bg = (const float*)d_in[6];
    const float* Wo = (const float*)d_in[7];
    const float* bo = (const float*)d_in[8];
    float* out = (float*)d_out;

    char* ws = (char*)d_ws;
    unsigned short* xbf = (unsigned short*)(ws + 0);         // 50331648 B
    unsigned short* Wx  = (unsigned short*)(ws + 50331648);  //  4718592 B
    unsigned short* Wh  = (unsigned short*)(ws + 55050240);  //  4718592 B
    unsigned short* hb0 = (unsigned short*)(ws + 59768832);  //    98304 B
    unsigned short* hb1 = (unsigned short*)(ws + 59867136);  //    98304 B
    unsigned*       bar = (unsigned*)      (ws + 59965440);  //     2048 B
    float*          Zx  = (float*)         (ws + 59967488);  // 402653184 B
    (void)in_sizes; (void)n_in; (void)out_size; (void)ws_size;

    {
        int n = ROWS * DD;
        cvt_x_kernel<<<(n + 255) / 256, 256, 0, stream>>>(x, xbf, n);
    }
    {
        int n = 4 * HH * DH;
        repack_w_kernel<<<(n + 255) / 256, 256, 0, stream>>>(Wf, Wi, Wg, Wo, Wx, Wh);
    }
    zero_bar_kernel<<<2, 256, 0, stream>>>(bar, TT);

    {
        dim3 grid(G4 / 128, ROWS / 64);
        xproj_gemm_kernel<<<grid, 256, 0, stream>>>(xbf, Wx, Zx);
    }

    // Persistent recurrent kernel: 24 WGs, 288KB dynamic LDS each
    lstm_persistent_kernel<<<NWG, 256, (4 * 32 * HH + BB * HH) * 2, stream>>>(
        Zx, Wh, bf, bi, bg, bo, hb0, hb1, bar, out);
}
